// ELECTRALSTMClassification_15899968929863
// MI455X (gfx1250) — matrix operand order repaired and checked by reading, not compiled, not run on hardware
//
#include <hip/hip_runtime.h>
#include <hip/hip_bf16.h>
#include <math.h>

// Problem constants (from reference): B=64, S=512, H=768, L=8, NL=10
#define B_   64
#define S_   512
#define H_   768
#define L_   8
#define G4   3072   // 4*H
#define BL_  512    // B*L
#define NL_  10
#define KS_  2      // K-split slices for the recurrent GEMM

typedef __bf16 bf16;
typedef __attribute__((ext_vector_type(16))) __bf16 v16bf;
typedef __attribute__((ext_vector_type(8)))  __bf16 v8bf;
typedef __attribute__((ext_vector_type(8)))  float  v8f;

// Pin the per-iteration schedule: issue the 10 next-iteration VMEM reads first,
// then the 4 WMMAs of the current iteration. This keeps two fragment sets live
// (defeating the scheduler's register-coalescing) so the waitcnt pass emits
// s_wait_loadcnt 10 (older loads only) instead of draining to 0 per WMMA.
#if __has_builtin(__builtin_amdgcn_sched_group_barrier)
#define SCHED_PIPELINE()                                      \
    do {                                                      \
        __builtin_amdgcn_sched_group_barrier(0x020, 10, 0);   \
        __builtin_amdgcn_sched_group_barrier(0x008, 4, 0);    \
    } while (0)
#else
#define SCHED_PIPELINE() do { } while (0)
#endif

// ---------------- WMMA fragment helpers (CDNA5 16x16x32 bf16 layouts) ----------------
// A (16x32, MxK): lanes 0-15 hold M=lane, K = {k0..k0+7, k0+16..k0+23};
//                 lanes 16-31 hold M=lane-16, K = {k0+8..k0+15, k0+24..k0+31}.
__device__ __forceinline__ v16bf load_a_frag(const bf16* A, int lda, int m0, int k0, int lane) {
    int m   = m0 + (lane & 15);
    int klo = k0 + ((lane & 16) >> 1);   // +8 for lanes >= 16
    const bf16* p = A + (long)m * lda + klo;
    v8bf lo = *(const v8bf*)(p);
    v8bf hi = *(const v8bf*)(p + 16);
    v16bf a;
#pragma unroll
    for (int j = 0; j < 8; ++j) { a[j] = lo[j]; a[j + 8] = hi[j]; }
    return a;
}

// B (32x16, KxN) built as W^T from row-major W (N x K):
// lanes 0-15: N=n0+lane, K = {k0..k0+7, k0+16..k0+23} ; lanes 16-31: N=n0+lane-16, K = {k0+8..k0+15, k0+24..k0+31}.
__device__ __forceinline__ v16bf load_b_fragT(const bf16* W, int ldw, int n0, int k0, int lane) {
    int n  = n0 + (lane & 15);
    int kk = k0 + ((lane & 16) >> 1);    // +8 for lanes >= 16
    const bf16* p = W + (long)n * ldw + kk;
    v8bf lo = *(const v8bf*)(p);
    v8bf hi = *(const v8bf*)(p + 16);
    v16bf b;
#pragma unroll
    for (int j = 0; j < 8; ++j) { b[j] = lo[j]; b[j + 8] = hi[j]; }
    return b;
}

// C/D (16x16 f32): VGPR v, lanes 0-15 -> (M=m0+v, N=n0+lane); lanes 16-31 -> (M=m0+8+v, N=n0+lane-16)
__device__ __forceinline__ void store_c_frag(float* C, int ldc, int m0, int n0, int lane, v8f acc) {
    int n     = n0 + (lane & 15);
    int mbase = m0 + ((lane & 16) >> 1);
#pragma unroll
    for (int v = 0; v < 8; ++v) C[(long)(mbase + v) * ldc + n] = acc[v];
}

#define WMMA_BF16(a, b, c) \
    __builtin_amdgcn_wmma_f32_16x16x32_bf16(false, (a), false, (b), (short)0, (c), false, false)

// One 16-row x 64-col output strip, K range [kbeg,kend). Software-pipelined with
// a pinned VMEM->WMMA group schedule per iteration.
__device__ __forceinline__ void gemm_strip_16x64(const bf16* __restrict__ A,
                                                 const bf16* __restrict__ W,
                                                 float* __restrict__ C, int ldc,
                                                 int m0, int n0, int kbeg, int kend, int lane) {
    // one-shot prefetch of this wave's weight strip (lanes spread over the 64 rows)
    __builtin_prefetch((const void*)(W + (long)(n0 + lane * 2) * H_ + kbeg), 0, 3);

    v8f a0 = {}, a1 = {}, a2 = {}, a3 = {};
    v16bf af = load_a_frag(A, H_, m0, kbeg, lane);
    v16bf b0 = load_b_fragT(W, H_, n0,      kbeg, lane);
    v16bf b1 = load_b_fragT(W, H_, n0 + 16, kbeg, lane);
    v16bf b2 = load_b_fragT(W, H_, n0 + 32, kbeg, lane);
    v16bf b3 = load_b_fragT(W, H_, n0 + 48, kbeg, lane);
    for (int k0 = kbeg; k0 + 32 < kend; k0 += 32) {
        v16bf afn = load_a_frag(A, H_, m0, k0 + 32, lane);
        v16bf b0n = load_b_fragT(W, H_, n0,      k0 + 32, lane);
        v16bf b1n = load_b_fragT(W, H_, n0 + 16, k0 + 32, lane);
        v16bf b2n = load_b_fragT(W, H_, n0 + 32, k0 + 32, lane);
        v16bf b3n = load_b_fragT(W, H_, n0 + 48, k0 + 32, lane);
        a0 = WMMA_BF16(af, b0, a0);
        a1 = WMMA_BF16(af, b1, a1);
        a2 = WMMA_BF16(af, b2, a2);
        a3 = WMMA_BF16(af, b3, a3);
        af = afn; b0 = b0n; b1 = b1n; b2 = b2n; b3 = b3n;
        SCHED_PIPELINE();
    }
    a0 = WMMA_BF16(af, b0, a0);
    a1 = WMMA_BF16(af, b1, a1);
    a2 = WMMA_BF16(af, b2, a2);
    a3 = WMMA_BF16(af, b3, a3);
    store_c_frag(C, ldc, m0, n0,      lane, a0);
    store_c_frag(C, ldc, m0, n0 + 16, lane, a1);
    store_c_frag(C, ldc, m0, n0 + 32, lane, a2);
    store_c_frag(C, ldc, m0, n0 + 48, lane, a3);
}

__device__ __forceinline__ float sigf(float x)  { return 1.0f / (1.0f + expf(-x)); }
__device__ __forceinline__ float geluf(float x) { return 0.5f * x * (1.0f + erff(x * 0.70710678118654752440f)); }

// ---------------- 1) fp32 -> bf16 weight conversion ----------------
__global__ __launch_bounds__(256) void cvt_f32_bf16(const float* __restrict__ src,
                                                    bf16* __restrict__ dst, int n) {
    int i = blockIdx.x * blockDim.x + threadIdx.x;
    if (i < n) dst[i] = (bf16)src[i];
}

// ---------------- 2) gather sep rows (forward + time-reversed copies) ----------------
__global__ __launch_bounds__(256) void gather_sep(const float* __restrict__ electra,
                                                  const int* __restrict__ sep_bx,
                                                  const int* __restrict__ sep_by,
                                                  bf16* __restrict__ sep_f,
                                                  bf16* __restrict__ sep_r) {
    int idx = blockIdx.x * blockDim.x + threadIdx.x;   // over BL_*H_
    if (idx >= BL_ * H_) return;
    int r = idx / H_, i = idx - r * H_;
    int b = sep_bx[r];
    int t = r - b * L_;               // position within segment (sep_bx is sorted blocks)
    int s = sep_by[r];
    float v = electra[((long)b * S_ + s) * H_ + i];
    sep_f[((long)(b * L_ + t)) * H_ + i]             = (bf16)v;
    sep_r[((long)(b * L_ + (L_ - 1 - t))) * H_ + i]  = (bf16)v;
}

// ---------------- 3) zero-init LSTM state ----------------
__global__ __launch_bounds__(256) void init_state(float* __restrict__ cstate, bf16* __restrict__ hbf) {
    int i = blockIdx.x * blockDim.x + threadIdx.x;     // over 2*B_*H_
    if (i >= 2 * B_ * H_) return;
    cstate[i] = 0.0f;
    hbf[i]    = (bf16)0.0f;
}

// ---------------- 4) input-gate GEMM: xg[dir] = sep[dir] (512x768) * w_ih[dir]^T (768x3072) ----------------
// Exact grid: 3072 waves (2 dirs x 32 row tiles x 48 column groups of 64).
__global__ __launch_bounds__(256) void xg_gemm(const bf16* __restrict__ sepf, const bf16* __restrict__ sepr,
                                               const bf16* __restrict__ wihf, const bf16* __restrict__ wihb,
                                               float* __restrict__ xgf, float* __restrict__ xgb) {
    int gwave = (blockIdx.x * blockDim.x + threadIdx.x) >> 5;
    int lane  = threadIdx.x & 31;
    int tg  = gwave % 48;            // 48 groups of 64 columns = 3072
    int tmp = gwave / 48;
    int tm  = tmp % 32;              // 32 row tiles = 512 rows
    int dir = tmp / 32;              // 2 directions
    const bf16* A = dir ? sepr : sepf;
    const bf16* W = dir ? wihb : wihf;
    float*      C = dir ? xgb  : xgf;
    gemm_strip_16x64(A, W, C, G4, tm * 16, tg * 64, 0, H_, lane);
}

// ---------------- 5) recurrent GEMM (K-split): gg[dir][ks] = h[dir] * w_hh[dir]^T over K slice ----------------
// Exact grid: 768 waves (2 dirs x 2 K-slices x 4 row tiles x 48 column groups).
__global__ __launch_bounds__(256) void rec_gemm(const bf16* __restrict__ hbf,
                                                const bf16* __restrict__ whhf, const bf16* __restrict__ whhb,
                                                float* __restrict__ gg) {
    int gwave = (blockIdx.x * blockDim.x + threadIdx.x) >> 5;
    int lane  = threadIdx.x & 31;
    int tg  = gwave % 48;
    int tmp = gwave / 48;
    int tm  = tmp % 4;               // 4 row tiles = 64 rows
    tmp /= 4;
    int ks  = tmp % KS_;             // K slice
    int dir = tmp / KS_;
    const bf16* A = hbf + (long)dir * B_ * H_;
    const bf16* W = dir ? whhb : whhf;
    float*      C = gg + ((long)dir * KS_ + ks) * B_ * G4;
    const int kw = H_ / KS_;         // 384
    gemm_strip_16x64(A, W, C, G4, tm * 16, tg * 64, ks * kw, (ks + 1) * kw, lane);
}

// ---------------- 6) LSTM gates: c' = sig(f)*c + sig(i)*tanh(g); h = sig(o)*tanh(c') ----------------
__global__ __launch_bounds__(256) void lstm_gate(const float* __restrict__ xgf, const float* __restrict__ xgb,
                                                 const float* __restrict__ gg,
                                                 const float* __restrict__ bihf, const float* __restrict__ bhhf,
                                                 const float* __restrict__ bihb, const float* __restrict__ bhhb,
                                                 float* __restrict__ cstate, bf16* __restrict__ hbf,
                                                 float* __restrict__ hbuf_f, float* __restrict__ hbuf_b, int t) {
    int idx = blockIdx.x * blockDim.x + threadIdx.x;   // over 2*B_*H_
    if (idx >= 2 * B_ * H_) return;
    int dir = idx / (B_ * H_);
    int rem = idx - dir * (B_ * H_);
    int b = rem / H_, j = rem - b * H_;
    const float* xg = (dir ? xgb : xgf) + (long)(b * L_ + t) * G4;
    const float* g0 = gg + ((long)dir * KS_ + 0) * B_ * G4 + (long)b * G4;
    const float* g1 = gg + ((long)dir * KS_ + 1) * B_ * G4 + (long)b * G4;
    const float* bi = dir ? bihb : bihf;
    const float* bh = dir ? bhhb : bhhf;
    float gi = xg[j]          + g0[j]          + g1[j]          + bi[j]          + bh[j];
    float gf = xg[H_ + j]     + g0[H_ + j]     + g1[H_ + j]     + bi[H_ + j]     + bh[H_ + j];
    float gc = xg[2 * H_ + j] + g0[2 * H_ + j] + g1[2 * H_ + j] + bi[2 * H_ + j] + bh[2 * H_ + j];
    float go = xg[3 * H_ + j] + g0[3 * H_ + j] + g1[3 * H_ + j] + bi[3 * H_ + j] + bh[3 * H_ + j];
    float cp = cstate[idx];
    float c  = sigf(gf) * cp + sigf(gi) * tanhf(gc);
    float h  = sigf(go) * tanhf(c);
    cstate[idx] = c;
    hbf[idx]    = (bf16)h;
    if (dir == 0) hbuf_f[(long)(b * L_ + t) * H_ + j] = h;            // forward: time t
    else          hbuf_b[(long)(b * L_ + (L_ - 1 - t)) * H_ + j] = h; // backward: un-reversed time
}

// ---------------- 7) heads: gelu + fc1 (3840->10), fc2 on sep_first/sep_last (1536->10) ----------------
__global__ __launch_bounds__(128) void final_heads(const float* __restrict__ electra,
                                                   const float* __restrict__ hbuf_f,
                                                   const float* __restrict__ hbuf_b,
                                                   const float* __restrict__ fc1_w, const float* __restrict__ fc1_b,
                                                   const float* __restrict__ fc2_w, const float* __restrict__ fc2_b,
                                                   float* __restrict__ out) {
    __shared__ float xact[5 * H_];   // gelu(concat[cls, sep_first, sep_last]) : 3840
    __shared__ float sf[2 * H_];     // raw sep_first : 1536
    __shared__ float sl[2 * H_];     // raw sep_last  : 1536
    int b   = blockIdx.x;
    int tid = threadIdx.x;
    for (int i = tid; i < H_; i += 128) {
        float v = electra[((long)b * S_) * H_ + i];   // cls = electra[b, 0, :]
        xact[i] = geluf(v);
    }
    for (int i = tid; i < 2 * H_; i += 128) {
        float vf = (i < H_) ? hbuf_f[(long)(b * L_) * H_ + i]
                            : hbuf_b[(long)(b * L_) * H_ + (i - H_)];
        sf[i] = vf;  xact[H_ + i] = geluf(vf);
        float vl = (i < H_) ? hbuf_f[(long)(b * L_ + (L_ - 1)) * H_ + i]
                            : hbuf_b[(long)(b * L_ + (L_ - 1)) * H_ + (i - H_)];
        sl[i] = vl;  xact[3 * H_ + i] = geluf(vl);
    }
    __syncthreads();
    int lane = tid & 31, wave = tid >> 5;
    for (int o = wave; o < 3 * NL_; o += 4) {
        int head = o / NL_, c = o - head * NL_;
        const float* vec; const float* w; float bias; int len;
        if (head == 0)      { vec = xact; w = fc1_w + c * (5 * H_); bias = fc1_b[c]; len = 5 * H_; }
        else if (head == 1) { vec = sf;   w = fc2_w + c * (2 * H_); bias = fc2_b[c]; len = 2 * H_; }
        else                { vec = sl;   w = fc2_w + c * (2 * H_); bias = fc2_b[c]; len = 2 * H_; }
        float p = 0.0f;
        for (int j = lane; j < len; j += 32) p += vec[j] * w[j];
#pragma unroll
        for (int off = 16; off; off >>= 1) p += __shfl_xor(p, off, 32);
        if (lane == 0) out[head * (B_ * NL_) + b * NL_ + c] = p + bias;
    }
}

// ---------------- host launch ----------------
extern "C" void kernel_launch(void* const* d_in, const int* in_sizes, int n_in,
                              void* d_out, int out_size, void* d_ws, size_t ws_size,
                              hipStream_t stream) {
    const float* electra = (const float*)d_in[0];
    const int*   sep_bx  = (const int*)d_in[1];
    const int*   sep_by  = (const int*)d_in[2];
    const float* w_ih_f  = (const float*)d_in[3];
    const float* w_hh_f  = (const float*)d_in[4];
    const float* b_ih_f  = (const float*)d_in[5];
    const float* b_hh_f  = (const float*)d_in[6];
    const float* w_ih_b  = (const float*)d_in[7];
    const float* w_hh_b  = (const float*)d_in[8];
    const float* b_ih_b  = (const float*)d_in[9];
    const float* b_hh_b  = (const float*)d_in[10];
    const float* fc1_w   = (const float*)d_in[11];
    const float* fc1_b   = (const float*)d_in[12];
    const float* fc2_w   = (const float*)d_in[13];
    const float* fc2_b   = (const float*)d_in[14];
    float* out = (float*)d_out;

    char* p = (char*)d_ws;
    auto alloc = [&](size_t bytes) -> void* {
        void* r = (void*)p;
        p += (bytes + 255) & ~((size_t)255);
        return r;
    };
    const int NW = G4 * H_;                       // 2,359,296 weight elements per matrix
    bf16* wihf_bf = (bf16*)alloc((size_t)NW * 2);
    bf16* whhf_bf = (bf16*)alloc((size_t)NW * 2);
    bf16* wihb_bf = (bf16*)alloc((size_t)NW * 2);
    bf16* whhb_bf = (bf16*)alloc((size_t)NW * 2);
    bf16* sep_f   = (bf16*)alloc((size_t)BL_ * H_ * 2);
    bf16* sep_r   = (bf16*)alloc((size_t)BL_ * H_ * 2);
    float* xg_f   = (float*)alloc((size_t)BL_ * G4 * 4);
    float* xg_b   = (float*)alloc((size_t)BL_ * G4 * 4);
    float* gg     = (float*)alloc((size_t)2 * KS_ * B_ * G4 * 4);
    float* cstate = (float*)alloc((size_t)2 * B_ * H_ * 4);
    bf16*  hbf    = (bf16*)alloc((size_t)2 * B_ * H_ * 2);
    float* hbuf_f = (float*)alloc((size_t)B_ * L_ * H_ * 4);
    float* hbuf_b = (float*)alloc((size_t)B_ * L_ * H_ * 4);

    // 1) weights -> bf16
    cvt_f32_bf16<<<(NW + 255) / 256, 256, 0, stream>>>(w_ih_f, wihf_bf, NW);
    cvt_f32_bf16<<<(NW + 255) / 256, 256, 0, stream>>>(w_hh_f, whhf_bf, NW);
    cvt_f32_bf16<<<(NW + 255) / 256, 256, 0, stream>>>(w_ih_b, wihb_bf, NW);
    cvt_f32_bf16<<<(NW + 255) / 256, 256, 0, stream>>>(w_hh_b, whhb_bf, NW);

    // 2) gather + 3) state init
    gather_sep<<<(BL_ * H_ + 255) / 256, 256, 0, stream>>>(electra, sep_bx, sep_by, sep_f, sep_r);
    init_state<<<(2 * B_ * H_ + 255) / 256, 256, 0, stream>>>(cstate, hbf);

    // 4) input-gate GEMM, both directions: 3072 waves -> 384 blocks x 256 (exact grid)
    xg_gemm<<<384, 256, 0, stream>>>(sep_f, sep_r, wihf_bf, wihb_bf, xg_f, xg_b);

    // 5/6) eight sequential LSTM steps (both directions, K-split x2, in each launch)
    for (int t = 0; t < L_; ++t) {
        rec_gemm<<<96, 256, 0, stream>>>(hbf, whhf_bf, whhb_bf, gg);   // 768 waves exact
        lstm_gate<<<(2 * B_ * H_ + 255) / 256, 256, 0, stream>>>(
            xg_f, xg_b, gg, b_ih_f, b_hh_f, b_ih_b, b_hh_b,
            cstate, hbf, hbuf_f, hbuf_b, t);
    }

    // 7) heads
    final_heads<<<B_, 128, 0, stream>>>(electra, hbuf_f, hbuf_b, fc1_w, fc1_b, fc2_w, fc2_b, out);
}
